// FusionLayerFeatsModule_71708773974455
// MI455X (gfx1250) — compile-verified, matrix operand order checked by reading, not run on hardware
//
#include <hip/hip_runtime.h>
#include <hip/hip_bf16.h>
#include <math.h>

// Problem constants (match reference)
#define TOK_N   131072
#define DIM_D   256
#define DIM_E   512
#define NBATCH  16
#define NLAYER  64
#define NSEG    (NBATCH * NLAYER)   // 1024
#define TD      (3 * DIM_D)         // 768
#define KC      (2 * DIM_D)         // 512
#define CPB     16                  // chunks per batch in pooling
#define FROWS   64                  // tokens per block in final GEMM (4 M-tiles)

typedef __attribute__((ext_vector_type(16))) _Float16 v16h;
typedef __attribute__((ext_vector_type(4)))  _Float16 v4h;
typedef __attribute__((ext_vector_type(8)))  float    v8f;

// ---------------------------------------------------------------- helpers
__device__ inline void atomicMaxF(float* addr, float v) {
  // IEEE ordered-int trick; addr must be initialized to -inf
  if (v >= 0.0f) atomicMax((int*)addr, __float_as_int(v));
  else           atomicMin((unsigned int*)addr, __float_as_uint(v));
}

__device__ inline float gelu_exact(float x) {
  return 0.5f * x * (1.0f + erff(x * 0.70710678118654752440f));
}

// Load one 16x32 f16 WMMA fragment (A-row or B-col) from a K-major array.
// ISA 7.12.2: lane<16 -> K {0..7, 16..23}; lane>=16 -> K {8..15, 24..31}.
__device__ inline v16h load_frag(const _Float16* base, int ldk, int rc, int kbase, int lane) {
  const _Float16* p = base + (size_t)rc * ldk + kbase + ((lane & 16) ? 8 : 0);
  v16h f;
#pragma unroll
  for (int e = 0; e < 8; ++e)  f[e] = p[e];
#pragma unroll
  for (int e = 8; e < 16; ++e) f[e] = p[e + 8];
  return f;
}

// ---------------------------------------------------------------- K0: init
__global__ __launch_bounds__(256) void init_ws_kernel(
    float* seg_sum, float* seg_max, float* seg_attn, float* seg_cnt,
    float* bmax, float* bz) {
  int i = blockIdx.x * 256 + threadIdx.x;
  if (i < NSEG * DIM_D) {
    seg_sum[i]  = 0.0f;
    seg_attn[i] = 0.0f;
    seg_max[i]  = -INFINITY;
  }
  if (i < NSEG)   seg_cnt[i] = 0.0f;
  if (i < NBATCH) { bmax[i] = -INFINITY; bz[i] = 0.0f; }
}

// K0b: W[k][n] fp32 (row-major K x Nc) -> WT[n][k] f16 (K-major per column)
__global__ __launch_bounds__(256) void transpose_f16_kernel(
    const float* __restrict__ W, _Float16* __restrict__ WT, int K, int Nc) {
  int i = blockIdx.x * 256 + threadIdx.x;
  if (i >= K * Nc) return;
  int k = i / Nc, n = i % Nc;
  WT[(size_t)n * K + k] = (_Float16)W[i];
}

// ---------------------------------------------------------------- K1: logits
// one wave per token: logits = feats @ Wa + ba ; seg id ; per-batch max
__global__ __launch_bounds__(256) void logits_kernel(
    const float* __restrict__ feats, const float* __restrict__ Wa,
    const float* __restrict__ ba, const int* __restrict__ cu,
    const int* __restrict__ layer_ids, float* logits, int* seg, float* bmax) {
  int warp = threadIdx.x >> 5, lane = threadIdx.x & 31;
  int t = blockIdx.x * 8 + warp;
  float p = 0.0f;
#pragma unroll
  for (int k = 0; k < 8; ++k) {
    int d = k * 32 + lane;
    p += feats[(size_t)t * DIM_D + d] * Wa[d];
  }
#pragma unroll
  for (int off = 16; off > 0; off >>= 1) p += __shfl_xor(p, off, 32);
  if (lane == 0) {
    float lg = p + ba[0];
    logits[t] = lg;
    int lo = 0, hi = NBATCH - 1;           // batch id via binary search
    while (lo < hi) { int mid = (lo + hi + 1) >> 1; if (t >= cu[mid]) lo = mid; else hi = mid - 1; }
    seg[t] = lo * NLAYER + layer_ids[t];
    atomicMaxF(&bmax[lo], lg);
  }
}

// ---------------------------------------------------------------- K2: exp + Z
__global__ __launch_bounds__(256) void softmax_z_kernel(
    float* logits, const int* __restrict__ seg,
    const float* __restrict__ bmax, float* bz) {
  __shared__ float zloc[NBATCH];
  if (threadIdx.x < NBATCH) zloc[threadIdx.x] = 0.0f;
  __syncthreads();
  int t = blockIdx.x * 256 + threadIdx.x;
  int b = seg[t] >> 6;
  float e = expf(logits[t] - bmax[b]);
  logits[t] = e;                        // reuse buffer: now holds e
  atomicAdd(&zloc[b], e);
  __syncthreads();
  if (threadIdx.x < NBATCH && zloc[threadIdx.x] != 0.0f)
    atomicAdd(&bz[threadIdx.x], zloc[threadIdx.x]);
}

// ---------------------------------------------------------------- K3: pooling
// grid = NBATCH * CPB * 4 ; block owns (batch-chunk, 64-wide D slice),
// privatizes all 64 layers in LDS, merges with global atomics.
__global__ __launch_bounds__(256) void pool_kernel(
    const float* __restrict__ feats, const int* __restrict__ cu,
    const int* __restrict__ layer_ids, const float* __restrict__ evals,
    const float* __restrict__ bz,
    float* g_sum, float* g_max, float* g_attn, float* g_cnt) {
  __shared__ float ssum[NLAYER * 64];
  __shared__ float smax[NLAYER * 64];
  __shared__ float sattn[NLAYER * 64];
  __shared__ float scnt[NLAYER];

  int bidx  = blockIdx.x;
  int slice = bidx & 3;
  int chunk = (bidx >> 2) & (CPB - 1);
  int b     = bidx >> 6;

  for (int i = threadIdx.x; i < NLAYER * 64; i += 256) {
    ssum[i] = 0.0f; sattn[i] = 0.0f; smax[i] = -INFINITY;
  }
  if (threadIdx.x < NLAYER) scnt[threadIdx.x] = 0.0f;
  __syncthreads();

  int s0 = cu[b], s1 = cu[b + 1];
  int len = s1 - s0;
  int start = s0 + (int)(((long long)chunk * len) / CPB);
  int end   = s0 + (int)(((long long)(chunk + 1) * len) / CPB);
  float invz = 1.0f / bz[b];

  int warp = threadIdx.x >> 5, lane = threadIdx.x & 31;
  for (int t = start + warp; t < end; t += 8) {
    int lid = layer_ids[t];
    float w = evals[t] * invz;
    int lbase = lid * 64;
#pragma unroll
    for (int k = 0; k < 2; ++k) {
      int dl = k * 32 + lane;
      float v = feats[(size_t)t * DIM_D + slice * 64 + dl];
      atomicAdd(&ssum[lbase + dl], v);
      atomicMaxF(&smax[lbase + dl], v);
      atomicAdd(&sattn[lbase + dl], v * w);
    }
    if (slice == 0 && lane == 0) atomicAdd(&scnt[lid], 1.0f);
  }
  __syncthreads();

  for (int i = threadIdx.x; i < NLAYER * 64; i += 256) {
    int lid = i >> 6, dl = i & 63;
    size_t g = (size_t)(b * NLAYER + lid) * DIM_D + slice * 64 + dl;
    atomicAdd(&g_sum[g], ssum[i]);
    atomicAdd(&g_attn[g], sattn[i]);
    float mv = smax[i];
    if (mv > -INFINITY) atomicMaxF(&g_max[g], mv);
  }
  if (slice == 0 && threadIdx.x < NLAYER && scnt[threadIdx.x] != 0.0f)
    atomicAdd(&g_cnt[b * NLAYER + threadIdx.x], scnt[threadIdx.x]);
}

// ---------------------------------------------------------------- K4a: pooled->f16
__global__ __launch_bounds__(256) void finalize_pool_kernel(
    const float* __restrict__ g_sum, const float* __restrict__ g_max,
    const float* __restrict__ g_attn, const float* __restrict__ g_cnt,
    _Float16* pooled16) {
  int i = blockIdx.x * 256 + threadIdx.x;
  if (i >= NSEG * DIM_D) return;
  int s = i >> 8, d = i & 255;
  float cnt = g_cnt[s];
  float avg = g_sum[i] / fmaxf(cnt, 1.0f);
  float mx  = (cnt > 0.0f) ? g_max[i] : 0.0f;
  size_t base = (size_t)s * TD;
  pooled16[base + d]             = (_Float16)avg;
  pooled16[base + DIM_D + d]     = (_Float16)mx;
  pooled16[base + 2 * DIM_D + d] = (_Float16)g_attn[i];
}

// ---------------------------------------------------------------- K4b: GEMM1 + GELU
// h = gelu(pooled[1024,768] @ W1 + b1) -> h16[1024,512]
__global__ __launch_bounds__(256) void gemm1_kernel(
    const _Float16* __restrict__ A, const _Float16* __restrict__ BT,
    const float* __restrict__ bias, _Float16* Hout) {
  __shared__ _Float16 Atile[16 * TD];
  int m0 = blockIdx.y * 16;
  for (int i = threadIdx.x; i < 16 * TD; i += 256)
    Atile[i] = A[(size_t)(m0 + i / TD) * TD + (i % TD)];
  __syncthreads();

  int lane = threadIdx.x & 31, warp = threadIdx.x >> 5;
  int c0 = blockIdx.x * 256 + warp * 32;
  int arow = lane & 15;
  int bcol = c0 + (lane & 15);
  v8f acc0 = {}; v8f acc1 = {};
  for (int kk = 0; kk < TD / 32; ++kk) {
    int kb = kk * 32;
    v16h a  = load_frag(Atile, TD, arow, kb, lane);
    v16h b0 = load_frag(BT, TD, bcol, kb, lane);
    v16h b1 = load_frag(BT, TD, bcol + 16, kb, lane);
    acc0 = __builtin_amdgcn_wmma_f32_16x16x32_f16(false, a, false, b0, (short)0, acc0, false, false);
    acc1 = __builtin_amdgcn_wmma_f32_16x16x32_f16(false, a, false, b1, (short)0, acc1, false, false);
  }
  int nloc = lane & 15, mb = (lane & 16) ? 8 : 0;
#pragma unroll
  for (int j = 0; j < 8; ++j) {
    int m = m0 + mb + j;
    int c = c0 + nloc;
    Hout[(size_t)m * DIM_E + c]      = (_Float16)gelu_exact(acc0[j] + bias[c]);
    Hout[(size_t)m * DIM_E + c + 16] = (_Float16)gelu_exact(acc1[j] + bias[c + 16]);
  }
}

// ---------------------------------------------------------------- K4c: GEMM2
// layer_feats16 = h[1024,512] @ W2 + b2 -> [1024,256]
__global__ __launch_bounds__(256) void gemm2_kernel(
    const _Float16* __restrict__ A, const _Float16* __restrict__ BT,
    const float* __restrict__ bias, _Float16* Lout) {
  __shared__ _Float16 Atile[16 * DIM_E];
  int m0 = blockIdx.y * 16;
  for (int i = threadIdx.x; i < 16 * DIM_E; i += 256)
    Atile[i] = A[(size_t)(m0 + i / DIM_E) * DIM_E + (i % DIM_E)];
  __syncthreads();

  int lane = threadIdx.x & 31, warp = threadIdx.x >> 5;
  int c0 = warp * 32;
  int arow = lane & 15;
  int bcol = c0 + (lane & 15);
  v8f acc0 = {}; v8f acc1 = {};
  for (int kk = 0; kk < DIM_E / 32; ++kk) {
    int kb = kk * 32;
    v16h a  = load_frag(Atile, DIM_E, arow, kb, lane);
    v16h b0 = load_frag(BT, DIM_E, bcol, kb, lane);
    v16h b1 = load_frag(BT, DIM_E, bcol + 16, kb, lane);
    acc0 = __builtin_amdgcn_wmma_f32_16x16x32_f16(false, a, false, b0, (short)0, acc0, false, false);
    acc1 = __builtin_amdgcn_wmma_f32_16x16x32_f16(false, a, false, b1, (short)0, acc1, false, false);
  }
  int nloc = lane & 15, mb = (lane & 16) ? 8 : 0;
#pragma unroll
  for (int j = 0; j < 8; ++j) {
    int m = m0 + mb + j;
    int c = c0 + nloc;
    Lout[(size_t)m * DIM_D + c]      = (_Float16)(acc0[j] + bias[c]);
    Lout[(size_t)m * DIM_D + c + 16] = (_Float16)(acc1[j] + bias[c + 16]);
  }
}

// ---------------------------------------------------------------- K5: final GEMM
// out[t] = [layer_feats[seg[t]] | feats[t]] @ Wc + bc   (N x 256, K = 512)
// 64 tokens per block (4 M-tiles); each wave loads B frags once per K-step
// and feeds 4 WMMA pairs -> 4x less L2 traffic for Wc fragments.
__global__ __launch_bounds__(256) void final_kernel(
    const float* __restrict__ feats, const int* __restrict__ seg,
    const _Float16* __restrict__ lf16, const _Float16* __restrict__ BT,
    const float* __restrict__ bias, float* out) {
  __shared__ _Float16 Atile[FROWS * KC];   // 64 KB
  int m0 = blockIdx.x * FROWS;

  // stage A: [gathered | feats] as f16, vectorized 4-wide
  for (int i = threadIdx.x; i < FROWS * (DIM_D / 4); i += 256) {
    int r = i >> 6;
    int d = (i & 63) << 2;
    int sg = seg[m0 + r];
    *(v4h*)&Atile[r * KC + d] = *(const v4h*)&lf16[(size_t)sg * DIM_D + d];
    float4 f = *(const float4*)&feats[(size_t)(m0 + r) * DIM_D + d];
    v4h h;
    h[0] = (_Float16)f.x; h[1] = (_Float16)f.y;
    h[2] = (_Float16)f.z; h[3] = (_Float16)f.w;
    *(v4h*)&Atile[r * KC + DIM_D + d] = h;
  }
  __syncthreads();

  int lane = threadIdx.x & 31, warp = threadIdx.x >> 5;
  int c0 = warp * 32;
  int arow = lane & 15;
  int bcol = c0 + (lane & 15);
  v8f acc[8] = {};
#pragma unroll 2
  for (int kk = 0; kk < KC / 32; ++kk) {
    int kb = kk * 32;
    v16h b0 = load_frag(BT, KC, bcol, kb, lane);
    v16h b1 = load_frag(BT, KC, bcol + 16, kb, lane);
#pragma unroll
    for (int mt = 0; mt < 4; ++mt) {
      v16h a = load_frag(Atile + mt * 16 * KC, KC, arow, kb, lane);
      acc[2 * mt]     = __builtin_amdgcn_wmma_f32_16x16x32_f16(false, a, false, b0, (short)0, acc[2 * mt], false, false);
      acc[2 * mt + 1] = __builtin_amdgcn_wmma_f32_16x16x32_f16(false, a, false, b1, (short)0, acc[2 * mt + 1], false, false);
    }
  }
  int nloc = lane & 15, mb = (lane & 16) ? 8 : 0;
  int c  = c0 + nloc;
  float bi0 = bias[c], bi1 = bias[c + 16];
#pragma unroll
  for (int mt = 0; mt < 4; ++mt) {
#pragma unroll
    for (int j = 0; j < 8; ++j) {
      int m = m0 + mt * 16 + mb + j;
      out[(size_t)m * DIM_D + c]      = acc[2 * mt][j] + bi0;
      out[(size_t)m * DIM_D + c + 16] = acc[2 * mt + 1][j] + bi1;
    }
  }
}

// ---------------------------------------------------------------- host
extern "C" void kernel_launch(void* const* d_in, const int* in_sizes, int n_in,
                              void* d_out, int out_size, void* d_ws, size_t ws_size,
                              hipStream_t stream) {
  const float* feats     = (const float*)d_in[0];
  const int*   cu        = (const int*)d_in[1];
  const int*   layer_ids = (const int*)d_in[2];
  const float* Wa        = (const float*)d_in[3];
  const float* ba        = (const float*)d_in[4];
  const float* W1        = (const float*)d_in[5];
  const float* b1        = (const float*)d_in[6];
  const float* W2        = (const float*)d_in[7];
  const float* b2        = (const float*)d_in[8];
  const float* Wc        = (const float*)d_in[9];
  const float* bc        = (const float*)d_in[10];
  float* out = (float*)d_out;

  char* ws = (char*)d_ws;
  size_t off = 0;
  auto take = [&](size_t bytes) -> char* {
    char* p = ws + off;
    off = (off + bytes + 255) & ~(size_t)255;
    return p;
  };
  float*     logits   = (float*)take((size_t)TOK_N * 4);
  int*       seg      = (int*)take((size_t)TOK_N * 4);
  float*     bmax     = (float*)take(NBATCH * 4);
  float*     bz       = (float*)take(NBATCH * 4);
  float*     seg_sum  = (float*)take((size_t)NSEG * DIM_D * 4);
  float*     seg_max  = (float*)take((size_t)NSEG * DIM_D * 4);
  float*     seg_attn = (float*)take((size_t)NSEG * DIM_D * 4);
  float*     seg_cnt  = (float*)take(NSEG * 4);
  _Float16*  pooled16 = (_Float16*)take((size_t)NSEG * TD * 2);
  _Float16*  h16      = (_Float16*)take((size_t)NSEG * DIM_E * 2);
  _Float16*  lf16     = (_Float16*)take((size_t)NSEG * DIM_D * 2);
  _Float16*  W1T      = (_Float16*)take((size_t)DIM_E * TD * 2);
  _Float16*  W2T      = (_Float16*)take((size_t)DIM_D * DIM_E * 2);
  _Float16*  WcT      = (_Float16*)take((size_t)DIM_D * KC * 2);

  // K0: init accumulators
  init_ws_kernel<<<(NSEG * DIM_D + 255) / 256, 256, 0, stream>>>(
      seg_sum, seg_max, seg_attn, seg_cnt, bmax, bz);
  // K0b: weights -> f16 col-major (K-major per output column)
  transpose_f16_kernel<<<(TD * DIM_E + 255) / 256, 256, 0, stream>>>(W1, W1T, TD, DIM_E);
  transpose_f16_kernel<<<(DIM_E * DIM_D + 255) / 256, 256, 0, stream>>>(W2, W2T, DIM_E, DIM_D);
  transpose_f16_kernel<<<(KC * DIM_D + 255) / 256, 256, 0, stream>>>(Wc, WcT, KC, DIM_D);
  // K1: logits + seg ids + per-batch max
  logits_kernel<<<TOK_N / 8, 256, 0, stream>>>(feats, Wa, ba, cu, layer_ids, logits, seg, bmax);
  // K2: e = exp(logit - m), Z per batch
  softmax_z_kernel<<<TOK_N / 256, 256, 0, stream>>>(logits, seg, bmax, bz);
  // K3: LDS-privatized segment pooling
  pool_kernel<<<NBATCH * CPB * 4, 256, 0, stream>>>(
      feats, cu, layer_ids, logits, bz, seg_sum, seg_max, seg_attn, seg_cnt);
  // K4a: pooled -> f16 [nseg, 768]
  finalize_pool_kernel<<<(NSEG * DIM_D + 255) / 256, 256, 0, stream>>>(
      seg_sum, seg_max, seg_attn, seg_cnt, pooled16);
  // K4b: GEMM1 + GELU -> h16
  gemm1_kernel<<<dim3(DIM_E / 256, NSEG / 16), 256, 0, stream>>>(pooled16, W1T, b1, h16);
  // K4c: GEMM2 -> layer_feats f16
  gemm2_kernel<<<dim3(1, NSEG / 16), 256, 0, stream>>>(h16, W2T, b2, lf16);
  // K5: gather + final GEMM -> out fp32
  final_kernel<<<TOK_N / FROWS, 256, 0, stream>>>(feats, seg, lf16, WcT, bc, out);
}
